// AttentionHead_66133906424109
// MI455X (gfx1250) — matrix operand order repaired and checked by reading, not compiled, not run on hardware
//
#include <hip/hip_runtime.h>
#include <hip/hip_bf16.h>

// ---------------------------------------------------------------------------
// CDNA5 (gfx1250) fused attention head, wave32 + v_wmma_f32_16x16x32_f16.
//   B=4, S=4096, D_in=512, D_out=64
// Pass 1: Q/K/V projections (fp32 -> f16, vectorized staging,
//         Q pre-scaled by log2(e)/sqrt(Sk), V stored transposed)
// Pass 2: flash attention, causal, base-2 online softmax, fp32 accumulation.
//         Row sums computed by WMMA against an all-ones fragment; causal
//         masking only on the single diagonal tile per wave.
// ---------------------------------------------------------------------------

typedef __attribute__((ext_vector_type(16))) _Float16 v16h;
typedef __attribute__((ext_vector_type(8)))  float    v8f;

union Frag16 { v16h v; uint4 q[2]; _Float16 h[16]; };
union Acc8   { v8f  v; float f[8]; };
union H8     { _Float16 h[8]; uint4 u; };

#define WMMA_F16(a, b, c) \
    __builtin_amdgcn_wmma_f32_16x16x32_f16(false, (a), false, (b), (short)0, (c), false, false)

// ===========================================================================
// Projection: Out[16 x 64] tile = X[16 x 512] * W[512 x 64], f16 output.
// grid.x = 16384/16 = 1024 blocks, 128 threads (4 waves; wave w -> cols w*16..)
// TRANSPOSE = 0 -> Out[(b*S+s)*64 + e]       (Q, K)
// TRANSPOSE = 1 -> Out[(b*64 + e)*4096 + s]  (V transposed for flash PV)
// ===========================================================================
template <int TRANSPOSE>
__global__ __launch_bounds__(128)
void proj_kernel(const float* __restrict__ X, const float* __restrict__ W,
                 _Float16* __restrict__ Out, float scale)
{
    __shared__ _Float16 Ah[16 * 64];   // activation tile, [m][k] row-major (2 KB)
    __shared__ _Float16 Wt[64 * 64];   // weight tile transposed, [n][k]     (8 KB)

    const int tid  = threadIdx.x;
    const int lane = tid & 31;
    const int wave = tid >> 5;
    const int l16  = lane & 15;
    const int hi   = (lane >= 16);
    const int ka   = hi ? 16 : 0;
    const int row0 = blockIdx.x * 16;

    v8f cacc = {};

    for (int k0 = 0; k0 < 512; k0 += 64) {
        // ---- stage A tile: 16x64 fp32 -> f16; 8 contiguous elems/thread ----
        {
            const int idx = tid * 8;             // 0..1016
            const int r = idx >> 6, kk = idx & 63;
            const float* src = X + (size_t)(row0 + r) * 512 + k0 + kk;
            float4 f0 = *(const float4*)(src);
            float4 f1 = *(const float4*)(src + 4);
            H8 p;
            p.h[0] = (_Float16)f0.x; p.h[1] = (_Float16)f0.y;
            p.h[2] = (_Float16)f0.z; p.h[3] = (_Float16)f0.w;
            p.h[4] = (_Float16)f1.x; p.h[5] = (_Float16)f1.y;
            p.h[6] = (_Float16)f1.z; p.h[7] = (_Float16)f1.w;
            *(uint4*)(Ah + r * 64 + kk) = p.u;   // one b128 LDS store
        }
        // ---- stage W tile transposed: W[k0+kk][n] -> Wt[n][kk] -------------
#pragma unroll
        for (int j = 0; j < 8; ++j) {
            const int idx = tid * 32 + j * 4;        // 0..4092
            const int kk = idx >> 6, n = idx & 63;
            float4 f = *(const float4*)(W + (size_t)(k0 + kk) * 64 + n);
            Wt[(n + 0) * 64 + kk] = (_Float16)f.x;
            Wt[(n + 1) * 64 + kk] = (_Float16)f.y;
            Wt[(n + 2) * 64 + kk] = (_Float16)f.z;
            Wt[(n + 3) * 64 + kk] = (_Float16)f.w;
        }
        __syncthreads();

#pragma unroll
        for (int dk = 0; dk < 64; dk += 32) {
            Frag16 a, b;
            const _Float16* arow = Ah + l16 * 64 + dk;
            a.q[0] = *(const uint4*)(arow + ka);
            a.q[1] = *(const uint4*)(arow + ka + 8);
            const _Float16* brow = Wt + (wave * 16 + l16) * 64 + dk + (hi ? 16 : 0);
            b.q[0] = *(const uint4*)(brow);
            b.q[1] = *(const uint4*)(brow + 8);
            cacc = WMMA_F16(a.v, b.v, cacc);
        }
        __syncthreads();
    }

    // C layout: vgpr r -> row r + (hi?8:0), col = lane%16
    Acc8 c; c.v = cacc;
    const int n = wave * 16 + l16;
#pragma unroll
    for (int r = 0; r < 8; ++r) {
        const int row = row0 + r + (hi ? 8 : 0);    // global bs-row
        const _Float16 val = (_Float16)(c.f[r] * scale);
        if (TRANSPOSE) {
            const int bb = row >> 12, s = row & 4095;
            Out[(((size_t)bb * 64 + n) << 12) + s] = val;
        } else {
            Out[(size_t)row * 64 + n] = val;
        }
    }
}

// ===========================================================================
// One 32-key flash tile: scores (4 wmma), online softmax, row-sum (1 wmma),
// PV (4 wmma). MASKED instantiation only for the single diagonal tile.
// ===========================================================================
template <bool MASKED>
__device__ __forceinline__ void attn_tile(
    int k0, int qbase, int l16, int hi,
    const Frag16 (&aq)[2],
    const _Float16* __restrict__ kBase,   // Kh + b*S*64
    const _Float16* __restrict__ vBase,   // Vt + b*64*4096
    _Float16* pw,                         // per-wave P staging in LDS
    Acc8 (&acc)[4], Acc8& lacc, float (&m)[8], const Frag16& ones)
{
    // ----- scores S = Q * K^T, two 16x16 tiles over 32 keys ---------------
    Acc8 s[2];
#pragma unroll
    for (int kc = 0; kc < 2; ++kc) {
        const _Float16* krow = kBase + (size_t)(k0 + kc * 16 + l16) * 64 + (hi ? 16 : 0);
        Frag16 bk0, bk1;
        bk0.q[0] = *(const uint4*)(krow);       bk0.q[1] = *(const uint4*)(krow + 8);
        bk1.q[0] = *(const uint4*)(krow + 32);  bk1.q[1] = *(const uint4*)(krow + 40);
        v8f cz = {};
        cz = WMMA_F16(aq[0].v, bk0.v, cz);
        cz = WMMA_F16(aq[1].v, bk1.v, cz);
        s[kc].v = cz;
    }

    // ----- (optional) causal mask + online softmax (base 2) ---------------
#pragma unroll
    for (int r = 0; r < 8; ++r) {
        if (MASKED) {
            const int qrow = qbase + r;
#pragma unroll
            for (int kc = 0; kc < 2; ++kc) {
                int k = k0 + kc * 16 + l16;
                if (k > qrow) s[kc].f[r] = -INFINITY;
            }
        }
        float v = fmaxf(s[0].f[r], s[1].f[r]);
        v = fmaxf(v, __shfl_xor(v, 1, 32));
        v = fmaxf(v, __shfl_xor(v, 2, 32));
        v = fmaxf(v, __shfl_xor(v, 4, 32));
        v = fmaxf(v, __shfl_xor(v, 8, 32));
        const float mn   = fmaxf(m[r], v);
        const float corr = exp2f(m[r] - mn);
        m[r] = mn;
#pragma unroll
        for (int kc = 0; kc < 2; ++kc) {
            float p = exp2f(s[kc].f[r] - mn);
            pw[(r + (hi ? 8 : 0)) * 32 + kc * 16 + l16] = (_Float16)p;  // C->A relayout
        }
#pragma unroll
        for (int nt = 0; nt < 4; ++nt) acc[nt].f[r] *= corr;
        lacc.f[r] *= corr;
    }

    // ----- P fragment (A layout) ------------------------------------------
    Frag16 ap;
    {
        const _Float16* prow = pw + l16 * 32;
        const int ka = hi ? 16 : 0;
        ap.q[0] = *(const uint4*)(prow + ka);
        ap.q[1] = *(const uint4*)(prow + ka + 8);
    }

    // ----- row sums: l += P * ones (replicated across lanes) --------------
    lacc.v = WMMA_F16(ap.v, ones.v, lacc.v);

    // ----- PV: O += P(16x32) * V(32x64) -----------------------------------
#pragma unroll
    for (int nt = 0; nt < 4; ++nt) {
        const _Float16* vrow = vBase + ((size_t)(nt * 16 + l16) << 12) + k0 + (hi ? 16 : 0);
        Frag16 bv;
        bv.q[0] = *(const uint4*)(vrow);
        bv.q[1] = *(const uint4*)(vrow + 8);
        acc[nt].v = WMMA_F16(ap.v, bv.v, acc[nt].v);
    }
}

// ===========================================================================
// Flash attention. grid = (S/64, B), 128 threads (4 waves).
// Wave w handles queries [q0 + 16w, q0 + 16w + 15]: ktFull unmasked 32-key
// tiles, then exactly one diagonal (masked) tile. Branches are wave-uniform.
// ===========================================================================
__global__ __launch_bounds__(128)
void flash_kernel(const _Float16* __restrict__ Qh,
                  const _Float16* __restrict__ Kh,
                  const _Float16* __restrict__ Vt,
                  float* __restrict__ Out)
{
    __shared__ _Float16 Qs[64 * 64];        // 8 KB query tile
    __shared__ _Float16 Ps[4][16 * 32];     // per-wave P staging (C->A layout)

    const int tid  = threadIdx.x;
    const int lane = tid & 31;
    const int wave = tid >> 5;
    const int l16  = lane & 15;
    const int hi   = (lane >= 16);
    const int b    = blockIdx.y;
    const int q0   = blockIdx.x * 64;
    const size_t baseQK = ((size_t)b << 12) * 64;   // b * S * 64
    const _Float16* kBase = Kh + baseQK;
    const _Float16* vBase = Vt + (((size_t)b * 64) << 12);

    // cooperative Q tile load (8 KB), fully unrolled b128 copies
    {
        const uint4* src = (const uint4*)(Qh + baseQK + (size_t)q0 * 64);
        uint4* dst = (uint4*)Qs;
#pragma unroll
        for (int i = 0; i < 4; ++i) dst[tid + 128 * i] = src[tid + 128 * i];
    }
    __syncthreads();

    Acc8 acc[4];
#pragma unroll
    for (int nt = 0; nt < 4; ++nt) { v8f z = {}; acc[nt].v = z; }
    Acc8 lacc; { v8f z = {}; lacc.v = z; }
    float m[8];
#pragma unroll
    for (int r = 0; r < 8; ++r) m[r] = -INFINITY;

    // all-ones B fragment for WMMA row sums
    Frag16 ones;
#pragma unroll
    for (int j = 0; j < 16; ++j) ones.h[j] = (_Float16)1.0f;

    // Q fragments (d-chunks 0 and 32), reused across the whole key loop
    Frag16 aq[2];
    {
        const _Float16* qrow = Qs + (wave * 16 + l16) * 64;
#pragma unroll
        for (int d = 0; d < 2; ++d) {
            int koff = d * 32 + (hi ? 16 : 0);
            aq[d].q[0] = *(const uint4*)(qrow + koff);
            aq[d].q[1] = *(const uint4*)(qrow + koff + 8);
        }
    }

    _Float16* pw = &Ps[wave][0];
    const int qw0   = q0 + wave * 16;        // first query row of this wave
    const int qbase = qw0 + (hi ? 8 : 0);
    const int ktFull = (qw0 + 1) >> 5;       // tiles fully below the diagonal

    for (int kt = 0; kt < ktFull; ++kt) {
        __builtin_prefetch(kBase + (size_t)(kt * 32 + 32) * 64, 0, 0);
        attn_tile<false>(kt * 32, qbase, l16, hi, aq, kBase, vBase, pw,
                         acc, lacc, m, ones);
    }
    // exactly one diagonal tile per 16-row wave tile
    attn_tile<true>(ktFull * 32, qbase, l16, hi, aq, kBase, vBase, pw,
                    acc, lacc, m, ones);

    // ----- epilogue: O / l ------------------------------------------------
#pragma unroll
    for (int r = 0; r < 8; ++r) {
        const float inv = 1.0f / lacc.f[r];
        const int q = qw0 + (hi ? 8 : 0) + r;
        float* orow = Out + (((size_t)b << 12) + q) * 64;
#pragma unroll
        for (int nt = 0; nt < 4; ++nt)
            orow[nt * 16 + l16] = acc[nt].f[r] * inv;
    }
}

// ===========================================================================
extern "C" void kernel_launch(void* const* d_in, const int* in_sizes, int n_in,
                              void* d_out, int out_size, void* d_ws, size_t ws_size,
                              hipStream_t stream)
{
    (void)in_sizes; (void)n_in; (void)out_size; (void)ws_size;
    const float* Xk = (const float*)d_in[0];
    const float* Xv = (const float*)d_in[1];
    const float* Xq = (const float*)d_in[2];
    const float* Wk = (const float*)d_in[3];
    const float* Wq = (const float*)d_in[4];
    const float* Wv = (const float*)d_in[5];

    const size_t tensorElems = (size_t)4 * 4096 * 64;   // 1 Mi halves each
    _Float16* Kh = (_Float16*)d_ws;                     // [B*S][64]
    _Float16* Vt = Kh + tensorElems;                    // [B][64][S]
    _Float16* Qh = Vt + tensorElems;                    // [B*S][64], pre-scaled

    // Q scale: (1/sqrt(4096)) * log2(e), so softmax can run in base 2.
    const float qscale = 0.015625f * 1.4426950408889634f;

    const int rowBlocks = (4 * 4096) / 16;              // 1024
    proj_kernel<0><<<dim3(rowBlocks), 128, 0, stream>>>(Xk, Wk, Kh, 1.0f);
    proj_kernel<1><<<dim3(rowBlocks), 128, 0, stream>>>(Xv, Wv, Vt, 1.0f);
    proj_kernel<0><<<dim3(rowBlocks), 128, 0, stream>>>(Xq, Wq, Qh, qscale);

    flash_kernel<<<dim3(4096 / 64, 4), 128, 0, stream>>>(Qh, Kh, Vt, (float*)d_out);
}
